// RGCNLayer_80831284511450
// MI455X (gfx1250) — compile-verified, matrix operand order checked by reading, not run on hardware
//
#include <hip/hip_runtime.h>

// RGCN layer for MI455X (gfx1250, wave32, WMMA).
//
// out = relu( scatter_add(Y[etype][src] -> dst) + X @ S^T )
// with Y[r] = X @ W[r] computed by v_wmma_f32_16x16x32_f16 (f32 accumulate).
//
// v3: store path de-branched — tile bounds are wave-uniform, so the 32 result
// stores per lane share one base address + immediate offsets (no per-element
// exec-mask dance). Operands remain pre-converted/packed f16 (b128 loads).

typedef __attribute__((ext_vector_type(16))) _Float16 v16h;
typedef __attribute__((ext_vector_type(8)))  _Float16 v8h;
typedef __attribute__((ext_vector_type(4)))  _Float16 v4h;
typedef __attribute__((ext_vector_type(8)))  float    v8f;

#define D 64  // D_IN == D_OUT

// ---------------- prep: X (f32) -> Xh (f16), row-major ----------------
__global__ void __launch_bounds__(256)
cvt_x_kernel(const float* __restrict__ X, _Float16* __restrict__ Xh, long n4)
{
  const long t = (long)blockIdx.x * blockDim.x + threadIdx.x;
  if (t >= n4) return;
  const float4 v = ((const float4*)X)[t];
  v4h h; h[0] = (_Float16)v.x; h[1] = (_Float16)v.y;
         h[2] = (_Float16)v.z; h[3] = (_Float16)v.w;
  *(v4h*)(Xh + t * 4) = h;
}

// ---- prep: pack all B matrices into WMMA B-operand register layout ----
// Element for (slot, tile t, khalf, lane, e):
//   col n = t*16 + lane%16 ; k = e + 8*((e>=8) + (lane>=16)) + 32*khalf
//   slot < nRel : W[slot][k][n]        (k-major)
//   slot == nRel: S[n][k]              (self-loop, transposed)
// Flat layout: Bops[slot*4096 + (t*2+khalf)*512 + lane*16 + e]
__global__ void __launch_bounds__(256)
pack_b_kernel(const float* __restrict__ W, const float* __restrict__ S,
              _Float16* __restrict__ Bops, int nRel)
{
  const int id = blockIdx.x * blockDim.x + threadIdx.x;
  const int total = (nRel + 1) * 4096;
  if (id >= total) return;
  const int e    = id & 15;
  const int lane = (id >> 4) & 31;
  const int half = (id >> 9) & 1;
  const int t    = (id >> 10) & 3;
  const int slot = id >> 12;
  const int n = t * 16 + (lane & 15);
  const int k = e + 8 * ((e >> 3) + (lane >> 4)) + 32 * half;
  const float v = (slot < nRel) ? W[(long)slot * D * D + k * D + n]
                                : S[(long)n * D + k];
  Bops[id] = (_Float16)v;
}

static __device__ __forceinline__ v16h load_op(const _Float16* p0,
                                               const _Float16* p1)
{
  const v8h a = *(const v8h*)p0;
  const v8h b = *(const v8h*)p1;
  v16h r;
#pragma unroll
  for (int i = 0; i < 8; ++i) { r[i] = a[i]; r[i + 8] = b[i]; }
  return r;
}

// ------------------------- WMMA GEMM -------------------------
// One wave: 32 rows (2 m-tiles) x 64 cols for one slot.
// Slots 0..nRel-1 -> Y[r]; slot nRel -> self-loop, written into `out`
// (which also initializes `out` ahead of the edge atomics).
__global__ void __launch_bounds__(256)
rgcn_gemm_wmma(const _Float16* __restrict__ Xh,    // [nNodes, 64] f16
               const _Float16* __restrict__ Bops,  // packed B operands
               float* __restrict__ Y,              // [nRel, nNodes, 64]
               float* __restrict__ out,            // [nNodes, 64]
               int nNodes, int nRel)
{
  const int lane = threadIdx.x & 31;
  const long waveId = (long)blockIdx.x * (blockDim.x >> 5) + (threadIdx.x >> 5);
  const int relSlots = nRel + 1;
  const int nGroups = (nNodes + 31) >> 5;         // 32 rows per wave
  const int grp  = (int)(waveId / relSlots);
  const int slot = (int)(waveId % relSlots);
  if (grp >= nGroups) return;                     // wave-uniform: EXEC stays full

  const bool selfLoop = (slot == nRel);
  const int col = lane & 15;
  const int hi  = lane >> 4;

  // B operands: 8 x (2 x b128) contiguous loads (L0/L2-resident, 72KB total)
  v16h b[4][2];
  const _Float16* bb = Bops + (long)slot * 4096 + (long)lane * 16;
#pragma unroll
  for (int t = 0; t < 4; ++t)
#pragma unroll
    for (int h = 0; h < 2; ++h) {
      const _Float16* p = bb + (t * 2 + h) * 512;
      b[t][h] = load_op(p, p + 8);
    }

  float* dstBase = selfLoop ? out : (Y + (long)slot * nNodes * D);

#pragma unroll
  for (int mt = 0; mt < 2; ++mt) {
    const int m0 = (grp * 2 + mt) * 16;
    if (m0 >= nNodes) break;                      // wave-uniform
    int arow = m0 + col; if (arow >= nNodes) arow = nNodes - 1;
    // A layout: lane holds row m0+lane%16; per k-half the lane's 16 halfs are
    // two contiguous 8-half runs at row*64 + hi*8 + 32*half (+16 for run 2).
    const _Float16* xr = Xh + (long)arow * D + hi * 8;
    v16h a0 = load_op(xr,      xr + 16);          // K = 0..31
    v16h a1 = load_op(xr + 32, xr + 48);          // K = 32..63

    v8f acc[4];
#pragma unroll
    for (int t = 0; t < 4; ++t) {
      v8f c = {};
      c = __builtin_amdgcn_wmma_f32_16x16x32_f16(false, a0, false, b[t][0],
                                                 (short)0, c, false, false);
      c = __builtin_amdgcn_wmma_f32_16x16x32_f16(false, a1, false, b[t][1],
                                                 (short)0, c, false, false);
      acc[t] = c;
    }

    // D layout: VGPR v -> row M = m0 + v + 8*hi, col N = t*16 + lane%16.
    if (m0 + 16 <= nNodes) {
      // Fast path (always taken when 16 | nNodes): single base address,
      // 32 stores with immediate offsets (v*256 + t*64 bytes).
      float* dp = dstBase + (long)(m0 + 8 * hi) * D + col;
#pragma unroll
      for (int v = 0; v < 8; ++v)
#pragma unroll
        for (int t = 0; t < 4; ++t)
          dp[v * D + t * 16] = acc[t][v];
    } else {
      // Ragged tail (nNodes % 16 != 0): guarded stores.
#pragma unroll
      for (int v = 0; v < 8; ++v) {
        const int m = m0 + v + 8 * hi;
        if (m < nNodes) {
          float* dp = dstBase + (long)m * D + col;
#pragma unroll
          for (int t = 0; t < 4; ++t) dp[t * 16] = acc[t][v];
        }
      }
    }
  }
}

// ---------------- edge scatter: gather Y row, atomic add ----------------
__global__ void __launch_bounds__(256)
rgcn_scatter(const int* __restrict__ srcI, const int* __restrict__ dstI,
             const int* __restrict__ et, const float* __restrict__ Y,
             float* __restrict__ out, int nEdges, int nNodes)
{
  const long t = (long)blockIdx.x * blockDim.x + threadIdx.x;
  const long e = t >> 5;
  if (e >= nEdges) return;
  const int c = ((int)t & 31) * 2;
  const int s = srcI[e];
  const int d = dstI[e];
  const int r = et[e];
  const float* yr = Y + ((long)r * nNodes + s) * D + c;
  const float v0 = yr[0];
  const float v1 = yr[1];
  float* o = out + (long)d * D + c;
  atomicAdd(o,     v0);
  atomicAdd(o + 1, v1);
}

// ---------------- fallback path (ws too small): no precompute ----------------
__global__ void __launch_bounds__(256)
selfloop_init_kernel(const float* __restrict__ X, const float* __restrict__ S,
                     float* __restrict__ out, int nNodes)
{
  const long t = (long)blockIdx.x * blockDim.x + threadIdx.x;
  const long n = t >> 6;
  if (n >= nNodes) return;
  const int j = (int)t & 63;
  const float* xr = X + n * D;
  const float* sr = S + (long)j * D;
  float acc = 0.f;
#pragma unroll
  for (int k = 0; k < D; ++k) acc = fmaf(xr[k], sr[k], acc);
  out[n * D + j] = acc;
}

__global__ void __launch_bounds__(256)
rgcn_edge_direct(const float* __restrict__ X, const int* __restrict__ srcI,
                 const int* __restrict__ dstI, const int* __restrict__ et,
                 const float* __restrict__ W, float* __restrict__ out, int nEdges)
{
  const long t = (long)blockIdx.x * blockDim.x + threadIdx.x;
  const long e = t >> 6;
  if (e >= nEdges) return;
  const int c = (int)t & 63;
  const int s = srcI[e];
  const int d = dstI[e];
  const int r = et[e];
  const float* xr = X + (long)s * D;
  const float* wc = W + (long)r * D * D + c;
  float acc = 0.f;
#pragma unroll
  for (int k = 0; k < D; ++k) acc = fmaf(xr[k], wc[(long)k * D], acc);
  atomicAdd(out + (long)d * D + c, acc);
}

// ---------------- relu (float4) ----------------
__global__ void __launch_bounds__(256)
relu4_kernel(float* __restrict__ out, long n4)
{
  const long t = (long)blockIdx.x * blockDim.x + threadIdx.x;
  if (t >= n4) return;
  float4 v = ((const float4*)out)[t];
  v.x = v.x > 0.f ? v.x : 0.f;
  v.y = v.y > 0.f ? v.y : 0.f;
  v.z = v.z > 0.f ? v.z : 0.f;
  v.w = v.w > 0.f ? v.w : 0.f;
  ((float4*)out)[t] = v;
}

static inline size_t align256(size_t x) { return (x + 255) & ~(size_t)255; }

extern "C" void kernel_launch(void* const* d_in, const int* in_sizes, int n_in,
                              void* d_out, int out_size, void* d_ws, size_t ws_size,
                              hipStream_t stream) {
  const float* x   = (const float*)d_in[0];   // [nNodes, 64]
  const int*   ei  = (const int*)  d_in[1];   // [2, nEdges]
  const int*   et  = (const int*)  d_in[2];   // [nEdges]
  const float* w   = (const float*)d_in[3];   // [nRel, 64, 64]
  const float* slw = (const float*)d_in[4];   // [64, 64]

  const int nNodes = in_sizes[0] / D;         // 100000
  const int nEdges = in_sizes[2];             // 1200000
  const int nRel   = in_sizes[3] / (D * D);   // 8
  const int* srcI = ei;
  const int* dstI = ei + nEdges;
  float* out = (float*)d_out;

  // ws layout: [ Y f32 | Xh f16 | Bops f16 ]
  const size_t yBytes  = (size_t)nRel * nNodes * D * sizeof(float);
  const size_t xhOff   = align256(yBytes);
  const size_t xhBytes = (size_t)nNodes * D * sizeof(_Float16);
  const size_t bOff    = align256(xhOff + xhBytes);
  const size_t bBytes  = (size_t)(nRel + 1) * 4096 * sizeof(_Float16);
  const size_t need    = bOff + bBytes;
  const bool pre = (ws_size >= need) && (nRel > 0);

  if (pre) {
    float*     Y    = (float*)d_ws;
    _Float16*  Xh   = (_Float16*)((char*)d_ws + xhOff);
    _Float16*  Bops = (_Float16*)((char*)d_ws + bOff);

    { // prep: X -> f16
      const long n4 = (long)nNodes * (D / 4);
      cvt_x_kernel<<<(int)((n4 + 255) / 256), 256, 0, stream>>>(x, Xh, n4);
    }
    { // prep: pack B operands (72KB)
      const int total = (nRel + 1) * 4096;
      pack_b_kernel<<<(total + 255) / 256, 256, 0, stream>>>(w, slw, Bops, nRel);
    }
    { // WMMA GEMMs: Y[r] = X@W_r ; out = X@S^T (init)
      const int nGroups = (nNodes + 31) / 32;
      const long waves  = (long)nGroups * (nRel + 1);
      const long blocks = (waves + 7) / 8;          // 8 waves / 256-thread block
      rgcn_gemm_wmma<<<(int)blocks, 256, 0, stream>>>(Xh, Bops, Y, out,
                                                      nNodes, nRel);
    }
    { // edge scatter (bandwidth/atomic bound, L2-resident out)
      const long threads = (long)nEdges * 32;
      rgcn_scatter<<<(int)((threads + 255) / 256), 256, 0, stream>>>(
          srcI, dstI, et, Y, out, nEdges, nNodes);
    }
  } else {
    { // init out with self-loop
      const long threads = (long)nNodes * D;
      selfloop_init_kernel<<<(int)((threads + 255) / 256), 256, 0, stream>>>(
          x, slw, out, nNodes);
    }
    { // direct per-edge messages
      const long threads = (long)nEdges * D;
      rgcn_edge_direct<<<(int)((threads + 255) / 256), 256, 0, stream>>>(
          x, srcI, dstI, et, w, out, nEdges);
    }
  }

  { // relu
    const long n4 = (long)nNodes * (D / 4);
    relu4_kernel<<<(int)((n4 + 255) / 256), 256, 0, stream>>>(out, n4);
  }
}
